// TSOANet_34591666601999
// MI455X (gfx1250) — compile-verified, matrix operand order checked by reading, not run on hardware
//
#include <hip/hip_runtime.h>
#include <hip/hip_bf16.h>
#include <math.h>

#define B_   32
#define T_   40
#define LM_  24
#define LD_  32
#define LP_  16
#define D_   256
#define BT_  (B_ * T_)
#define LTOT_ 73

typedef __attribute__((ext_vector_type(16))) __bf16 v16bf;
typedef __attribute__((ext_vector_type(8)))  float  v8f;

// ---------- helpers ----------
__device__ __forceinline__ unsigned short f2bf(float x) {
    unsigned int u = __builtin_bit_cast(unsigned int, x);
    u += 0x7FFFu + ((u >> 16) & 1u);          // round-to-nearest-even
    return (unsigned short)(u >> 16);
}

// Load one 16x32 bf16 WMMA operand fragment from row-major LDS.
// A-matrix: row = lane&15, K striped per ISA 16-bit A layout (pairs, +8*half).
// B-matrix uses the same striping with N taken from lane (row0 = N base).
__device__ __forceinline__ v16bf load_frag(const unsigned short* p, int row0,
                                           int col0, int rstride) {
    const int lane = threadIdx.x & 31;
    const int m = lane & 15;
    const int h = lane >> 4;
    union { v16bf v; unsigned int u[8]; } f;
    const unsigned short* base = p + (row0 + m) * rstride + col0 + 8 * h;
#pragma unroll
    for (int q = 0; q < 8; ++q) {
        int kb = (q < 4) ? (2 * q) : (16 + 2 * (q - 4));
        f.u[q] = *reinterpret_cast<const unsigned int*>(base + kb);
    }
    return f.v;
}

// ---------- stage 1: time feature MLP ----------
__global__ __launch_bounds__(256) void time_feat_kernel(
    const float* __restrict__ seq_t, const float* __restrict__ W_sel,
    const float* __restrict__ b_sel, const float* __restrict__ W_time,
    const float* __restrict__ b_time, float* __restrict__ tf) {
    int bt = blockIdx.x, d = threadIdx.x;
    __shared__ float h[64];
    float t = seq_t[bt];
    if (d < 64) {
        float x = t * W_sel[d] + b_sel[d];
        h[d] = 1.0f - tanhf(x * x);
    }
    __syncthreads();
    float acc = b_time[d];
#pragma unroll 4
    for (int j = 0; j < 64; ++j) acc += W_time[d * 64 + j] * h[j];
    tf[(long long)bt * D_ + d] = acc;
}

// ---------- stage 2: embedding gather * mask ----------
__global__ __launch_bounds__(256) void gather_kernel(
    const int* __restrict__ codes, const float* __restrict__ mask,
    const float* __restrict__ emb, float* __restrict__ out) {
    long long i = blockIdx.x;
    int d = threadIdx.x;
    int code = codes[i];
    float m = mask[i];
    out[i * D_ + d] = emb[(long long)code * D_ + d] * m;
}

// ---------- stage 3: time-gated attention; writes bf16 gated output ----------
__global__ __launch_bounds__(256) void gate_kernel(
    const float* __restrict__ in, unsigned short* __restrict__ out_bf,
    const float* __restrict__ tf, const float* __restrict__ Wg,
    const float* __restrict__ bg, int L, float scale) {
    int bt = blockIdx.x, d = threadIdx.x;
    __shared__ float s_tf[D_], s_ctx[D_], s_p[64];
    const float* o = in + (long long)bt * L * D_;
    unsigned short* ob = out_bf + (long long)bt * L * D_;
    s_tf[d] = tf[(long long)bt * D_ + d];
    __syncthreads();
    if (d < L) {
        float a = 0.f;
#pragma unroll 4
        for (int e = 0; e < D_; ++e) a += s_tf[e] * o[d * D_ + e];
        s_p[d] = a * scale;
    }
    __syncthreads();
    if (d == 0) {
        float mx = -1e30f;
        for (int l = 0; l < L; ++l) mx = fmaxf(mx, s_p[l]);
        float sm = 0.f;
        for (int l = 0; l < L; ++l) { s_p[l] = __expf(s_p[l] - mx); sm += s_p[l]; }
        float inv = 1.f / sm;
        for (int l = 0; l < L; ++l) s_p[l] *= inv;
    }
    __syncthreads();
    float ctx = 0.f;
    for (int l = 0; l < L; ++l) ctx += s_p[l] * o[l * D_ + d];
    s_ctx[d] = ctx;
    __syncthreads();
    float g = bg[d];
#pragma unroll 4
    for (int e = 0; e < D_; ++e) g += Wg[d * D_ + e] * s_ctx[e];
    g = 1.f / (1.f + __expf(-g));
    for (int l = 0; l < L; ++l) ob[l * D_ + d] = f2bf(o[l * D_ + d] * g);
}

// ---------- generic WMMA flash attention (bf16 in, f32 out) ----------
// Per group: O = softmax(scale * Q Kv^T) Kv ; Q is Nq x 256, Kv is Nkv x 256.
// Block = 2 waves, each wave owns one 16-row query tile; keys streamed in
// 32-token tiles through LDS via GLOBAL_LOAD_ASYNC_TO_LDS_B128.
__global__ __launch_bounds__(64) void flash_attn_kernel(
    const unsigned short* __restrict__ Qg, const unsigned short* __restrict__ KVg,
    float* __restrict__ Og, int Nq, int Nkv, long long qStride,
    long long kvStride, long long oStride, float scale) {
    __shared__ unsigned short Qs[2][16 * D_];   // per-wave Q tile (bf16)
    __shared__ unsigned short Ks[32 * D_];      // key tile, row-major
    __shared__ unsigned short Vt[D_ * 32];      // value tile, transposed [d][key]
    __shared__ unsigned short Ps[2][16 * 32];   // per-wave P staging

    const int group = blockIdx.y;
    const int wave = threadIdx.x >> 5;
    const int lane = threadIdx.x & 31;
    const int h = lane >> 4;
    const int qtile = blockIdx.x * 2 + wave;

    const unsigned short* Qp  = Qg  + (long long)group * qStride;
    const unsigned short* KVp = KVg + (long long)group * kvStride;
    float*                Op  = Og  + (long long)group * oStride;

    // load my Q tile (clamp rows so padded tiles stay in-bounds)
    for (int i = lane; i < 16 * D_; i += 32) {
        int r = i >> 8, c = i & 255;
        int qr = qtile * 16 + r;
        if (qr >= Nq) qr = Nq - 1;
        Qs[wave][i] = Qp[(long long)qr * D_ + c];
    }

    v8f acc[16];
    const v8f zero = {0.f, 0.f, 0.f, 0.f, 0.f, 0.f, 0.f, 0.f};
#pragma unroll
    for (int t = 0; t < 16; ++t) acc[t] = zero;
    float m_state[8], l_state[8];
#pragma unroll
    for (int r = 0; r < 8; ++r) { m_state[r] = -1e30f; l_state[r] = 0.f; }

    const int nkt = (Nkv + 31) / 32;
    for (int kt = 0; kt < nkt; ++kt) {
        __syncthreads();  // previous tile fully consumed
        // async copy key tile (bf16, row-major) global -> LDS, 16B per op
        for (int i = threadIdx.x; i < 32 * (D_ / 8); i += 64) {
            int r = i >> 5;              // key row within tile
            int c8 = (i & 31) * 8;       // 8-element column chunk
            int tok = kt * 32 + r;
            if (tok < Nkv) {
                unsigned lds = (unsigned)(size_t)(void*)&Ks[r * D_ + c8];
                unsigned long long ga =
                    (unsigned long long)(size_t)(KVp + (long long)tok * D_ + c8);
                asm volatile("global_load_async_to_lds_b128 %0, %1, off"
                             :: "v"(lds), "v"(ga) : "memory");
            } else {
                unsigned short* z = &Ks[r * D_ + c8];
#pragma unroll
                for (int j = 0; j < 8; ++j) z[j] = 0;
            }
        }
        asm volatile("s_wait_asynccnt 0x0" ::: "memory");
        __syncthreads();
        // build transposed V tile from Ks (LDS -> LDS)
        for (int i = threadIdx.x; i < 32 * D_; i += 64) {
            int r = i >> 8, c = i & 255;
            Vt[c * 32 + r] = Ks[i];
        }
        __syncthreads();

        // S = Q K^T over 8 D-chunks, two 16-col score tiles
        v8f c0 = zero, c1 = zero;
#pragma unroll
        for (int d = 0; d < 8; ++d) {
            v16bf a  = load_frag(Qs[wave], 0, d * 32, D_);
            v16bf b0 = load_frag(Ks, 0,  d * 32, D_);
            v16bf b1 = load_frag(Ks, 16, d * 32, D_);
            c0 = __builtin_amdgcn_wmma_f32_16x16x32_bf16(false, a, false, b0,
                                                         (short)0, c0, false, false);
            c1 = __builtin_amdgcn_wmma_f32_16x16x32_bf16(false, a, false, b1,
                                                         (short)0, c1, false, false);
        }

        // online softmax per row (row = r + 8h across 16 lanes)
        const int col0 = kt * 32 + (lane & 15);
        const int col1 = col0 + 16;
#pragma unroll
        for (int r = 0; r < 8; ++r) {
            float s0 = (col0 < Nkv) ? c0[r] * scale : -1e30f;
            float s1 = (col1 < Nkv) ? c1[r] * scale : -1e30f;
            float mx = fmaxf(s0, s1);
            mx = fmaxf(mx, __shfl_xor(mx, 1, 32));
            mx = fmaxf(mx, __shfl_xor(mx, 2, 32));
            mx = fmaxf(mx, __shfl_xor(mx, 4, 32));
            mx = fmaxf(mx, __shfl_xor(mx, 8, 32));
            float m_new = fmaxf(m_state[r], mx);
            float p0 = __expf(s0 - m_new);
            float p1 = __expf(s1 - m_new);
            float rs = p0 + p1;
            rs += __shfl_xor(rs, 1, 32);
            rs += __shfl_xor(rs, 2, 32);
            rs += __shfl_xor(rs, 4, 32);
            rs += __shfl_xor(rs, 8, 32);
            float corr = __expf(m_state[r] - m_new);
            l_state[r] = l_state[r] * corr + rs;
            m_state[r] = m_new;
#pragma unroll
            for (int t = 0; t < 16; ++t) acc[t][r] *= corr;
            int row = r + 8 * h;
            Ps[wave][row * 32 + (lane & 15)]      = f2bf(p0);
            Ps[wave][row * 32 + (lane & 15) + 16] = f2bf(p1);
        }
        __syncthreads();  // P staging visible; keeps waves in lockstep

        // O += P V  (16 output column tiles, K=32 keys)
        v16bf pa = load_frag(Ps[wave], 0, 0, 32);
#pragma unroll
        for (int t = 0; t < 16; ++t) {
            v16bf b = load_frag(Vt, t * 16, 0, 32);
            acc[t] = __builtin_amdgcn_wmma_f32_16x16x32_bf16(false, pa, false, b,
                                                             (short)0, acc[t],
                                                             false, false);
        }
    }

    // normalize + store valid rows
#pragma unroll
    for (int r = 0; r < 8; ++r) {
        int row = qtile * 16 + r + 8 * h;
        if (row < Nq) {
            float inv = 1.f / l_state[r];
#pragma unroll
            for (int t = 0; t < 16; ++t)
                Op[(long long)row * D_ + t * 16 + (lane & 15)] = acc[t][r] * inv;
        }
    }
}

// ---------- stage 5: concat [med|diag|proc|tf] -> bf16 merge input ----------
__global__ __launch_bounds__(256) void concat_kernel(
    const float* __restrict__ med, const float* __restrict__ diag,
    const float* __restrict__ proc, const float* __restrict__ tf,
    unsigned short* __restrict__ merge_bf) {
    long long i = blockIdx.x;   // BT * 73
    int d = threadIdx.x;
    long long bt = i / LTOT_;
    int l = (int)(i % LTOT_);
    float v;
    if (l < LM_)              v = med[(bt * LM_ + l) * D_ + d];
    else if (l < LM_ + LD_)   v = diag[(bt * LD_ + (l - LM_)) * D_ + d];
    else if (l < LTOT_ - 1)   v = proc[(bt * LP_ + (l - LM_ - LD_)) * D_ + d];
    else                      v = tf[bt * D_ + d];
    merge_bf[i * D_ + d] = f2bf(v);
}

// ---------- f32 -> bf16 convert ----------
__global__ __launch_bounds__(256) void cvt_bf16_kernel(
    const float* __restrict__ in, unsigned short* __restrict__ out) {
    long long i = (long long)blockIdx.x * 256 + threadIdx.x;
    out[i] = f2bf(in[i]);
}

// ---------- stage 8: sum over LM + bias + positional encoding ----------
__global__ __launch_bounds__(256) void final_kernel(
    const float* __restrict__ med_att, const float* __restrict__ bias_med,
    const int* __restrict__ input_len, float* __restrict__ out) {
    int bt = blockIdx.x, d = threadIdx.x;
    int b = bt / T_, t = bt % T_;
    float s = 0.f;
#pragma unroll 4
    for (int l = 0; l < LM_; ++l)
        s += med_att[((long long)bt * LM_ + l) * D_ + d];
    s += bias_med[d];
    if (t < input_len[b]) {
        int j2 = d & ~1;
        float inv = __expf(-(float)j2 * 9.210340371976184f / (float)D_);
        float ang = (float)t * inv;
        s += (d & 1) ? __cosf(ang) : __sinf(ang);
    }
    out[(long long)bt * D_ + d] = s;
}

extern "C" void kernel_launch(void* const* d_in, const int* in_sizes, int n_in,
                              void* d_out, int out_size, void* d_ws,
                              size_t ws_size, hipStream_t stream) {
    const int*   med_codes = (const int*)d_in[0];
    const int*   diag_codes = (const int*)d_in[1];
    const int*   proc_codes = (const int*)d_in[2];
    const float* med_mask  = (const float*)d_in[3];
    const float* diag_mask = (const float*)d_in[4];
    const float* proc_mask = (const float*)d_in[5];
    const float* seq_time  = (const float*)d_in[6];
    const int*   input_len = (const int*)d_in[7];
    const float* emb_med   = (const float*)d_in[8];
    const float* emb_diag  = (const float*)d_in[9];
    const float* emb_proc  = (const float*)d_in[10];
    const float* bias_med  = (const float*)d_in[11];
    const float* W_sel     = (const float*)d_in[12];
    const float* b_sel     = (const float*)d_in[13];
    const float* W_time    = (const float*)d_in[14];
    const float* b_time    = (const float*)d_in[15];
    const float* Wg_med    = (const float*)d_in[16];
    const float* bg_med    = (const float*)d_in[17];
    const float* Wg_diag   = (const float*)d_in[18];
    const float* bg_diag   = (const float*)d_in[19];
    const float* Wg_proc   = (const float*)d_in[20];
    const float* bg_proc   = (const float*)d_in[21];
    float* out = (float*)d_out;

    const float scale = 1.0f / sqrtf((float)D_ + 1e-7f);

    const long long nMed = (long long)BT_ * LM_ * D_;   // 7,864,320
    const long long nDia = (long long)BT_ * LD_ * D_;   // 10,485,760
    const long long nPro = (long long)BT_ * LP_ * D_;   // 5,242,880
    const long long nMrg = (long long)BT_ * LTOT_ * D_; // 23,920,640

    // workspace layout
    float* tf       = (float*)d_ws;                 // BT*D
    float* gmed     = tf       + (long long)BT_ * D_;
    float* gdiag    = gmed     + nMed;
    float* gproc    = gdiag    + nDia;
    float* vmed     = gproc    + nPro;
    float* vdiag    = vmed     + nMed;
    float* vproc    = vdiag    + nDia;
    float* mergeout = vproc    + nPro;
    float* medatt   = mergeout + nMrg;
    unsigned short* gmed_bf  = (unsigned short*)(medatt + nMed);
    unsigned short* gdiag_bf = gmed_bf  + nMed;
    unsigned short* gproc_bf = gdiag_bf + nDia;
    unsigned short* mergein_bf  = gproc_bf + nPro;
    unsigned short* mergeout_bf = mergein_bf + nMrg;

    // 1. time feature
    time_feat_kernel<<<BT_, 256, 0, stream>>>(seq_time, W_sel, b_sel, W_time,
                                              b_time, tf);
    // 2. embedding gathers (f32)
    gather_kernel<<<BT_ * LM_, 256, 0, stream>>>(med_codes, med_mask, emb_med, gmed);
    gather_kernel<<<BT_ * LD_, 256, 0, stream>>>(diag_codes, diag_mask, emb_diag, gdiag);
    gather_kernel<<<BT_ * LP_, 256, 0, stream>>>(proc_codes, proc_mask, emb_proc, gproc);
    // 3. time-gated attention -> bf16 staging buffers
    gate_kernel<<<BT_, 256, 0, stream>>>(gmed,  gmed_bf,  tf, Wg_med,  bg_med,  LM_, scale);
    gate_kernel<<<BT_, 256, 0, stream>>>(gdiag, gdiag_bf, tf, Wg_diag, bg_diag, LD_, scale);
    gate_kernel<<<BT_, 256, 0, stream>>>(gproc, gproc_bf, tf, Wg_proc, bg_proc, LP_, scale);
    // 4. visit self-attentions over T*L tokens per batch (WMMA flash)
    {
        int Nm = T_ * LM_, Nd = T_ * LD_, Np = T_ * LP_;
        flash_attn_kernel<<<dim3(Nm / 32, B_), 64, 0, stream>>>(
            gmed_bf, gmed_bf, vmed, Nm, Nm, (long long)Nm * D_,
            (long long)Nm * D_, (long long)Nm * D_, scale);
        flash_attn_kernel<<<dim3(Nd / 32, B_), 64, 0, stream>>>(
            gdiag_bf, gdiag_bf, vdiag, Nd, Nd, (long long)Nd * D_,
            (long long)Nd * D_, (long long)Nd * D_, scale);
        flash_attn_kernel<<<dim3(Np / 32, B_), 64, 0, stream>>>(
            gproc_bf, gproc_bf, vproc, Np, Np, (long long)Np * D_,
            (long long)Np * D_, (long long)Np * D_, scale);
    }
    // 5. concat to bf16 merge input
    concat_kernel<<<BT_ * LTOT_, 256, 0, stream>>>(vmed, vdiag, vproc, tf, mergein_bf);
    // 6. merge self-attention, 73 tokens per (b,t): 5 q-tiles -> 3 blocks
    flash_attn_kernel<<<dim3(3, BT_), 64, 0, stream>>>(
        mergein_bf, mergein_bf, mergeout, LTOT_, LTOT_, (long long)LTOT_ * D_,
        (long long)LTOT_ * D_, (long long)LTOT_ * D_, scale);
    // 7. bf16 staging of merge output, then med cross-attention (scale = 1)
    cvt_bf16_kernel<<<(unsigned)(nMrg / 256), 256, 0, stream>>>(mergeout, mergeout_bf);
    flash_attn_kernel<<<dim3(1, BT_), 64, 0, stream>>>(
        mergeout_bf, mergeout_bf, medatt, LM_, LTOT_, (long long)LTOT_ * D_,
        (long long)LTOT_ * D_, (long long)LM_ * D_, 1.0f);
    // 8. reduce + bias + positional encoding
    final_kernel<<<BT_, 256, 0, stream>>>(medatt, bias_med, input_len, out);
}
